// DiffusionModule2_53008486367372
// MI455X (gfx1250) — compile-verified, matrix operand order checked by reading, not run on hardware
//
#include <hip/hip_runtime.h>
#include <cstdint>
#include <cstddef>

// ---------------- problem constants ----------------
constexpr int Bz = 2;
constexpr int Lz = 512;
constexpr int Dz = 512;
constexpr int Hz = 16;
constexpr int Pz = 128;
constexpr int NLz = 2;
constexpr int DHz = Dz / Hz;      // 32
constexpr int FFz = 4 * Dz;       // 2048
constexpr float EPSz = 1e-5f;
constexpr float NEGz = -1e9f;

typedef _Float16 half_t;
typedef __attribute__((ext_vector_type(16))) _Float16 v16h;
typedef __attribute__((ext_vector_type(8)))  _Float16 v8h;
typedef __attribute__((ext_vector_type(8)))  float    v8f;

// ---------------- device helpers ----------------
__device__ __forceinline__ float siluf(float x) { return x / (1.f + __expf(-x)); }
__device__ __forceinline__ float geluf(float x) {
  const float c = 0.7978845608028654f; // sqrt(2/pi)
  float t = tanhf(c * (x + 0.044715f * x * x * x));
  return 0.5f * x * (1.f + t);
}

// 16x32 f16 A/B fragment per ISA 7.12.2: lane<16 holds K {kb..kb+7, kb+16..kb+23},
// kb = k0 + 8*(lane>=16). With f16 source, that is two contiguous 16B vectors.
__device__ __forceinline__ v16h ld_frag(const half_t* p) {
  v8h a = *(const v8h*)(p);
  v8h b = *(const v8h*)(p + 16);
  v16h x;
#pragma unroll
  for (int i = 0; i < 8; ++i) { x[i] = a[i]; x[8 + i] = b[i]; }
  return x;
}

__device__ __forceinline__ v8f wmma_f16(v16h a, v16h b, v8f c) {
  return __builtin_amdgcn_wmma_f32_16x16x32_f16(false, a, false, b, (short)0, c,
                                                false, false);
}

// ---------------- fp32 -> f16 conversion (weights / one-shot operands)
__global__ __launch_bounds__(256) void cvt16_kernel(const float* __restrict__ src,
                                                    half_t* __restrict__ dst, long n) {
  long i = (long)blockIdx.x * 256 + threadIdx.x;
  if (i < n) dst[i] = (half_t)src[i];
}

// ---------------- pair bias: pfb[b,h,l,m] = silu(pair_feat[b,l,m,:]) . pfb_w[h,:]
// also colscore[b,m,l] = mean_h pfb  (for top-k over rows per column)
__global__ __launch_bounds__(256) void pfb_kernel(const float* __restrict__ pair_feat,
                                                  const float* __restrict__ pfb_w,
                                                  float* __restrict__ pfb,
                                                  float* __restrict__ colscore) {
  __shared__ float w[Hz * Pz];
  int tid = threadIdx.x;
  for (int i = tid; i < Hz * Pz; i += 256) w[i] = pfb_w[i];
  __syncthreads();
  long flat = (long)blockIdx.x * 256 + tid;        // b*L*L + l*L + m
  int b = (int)(flat / ((long)Lz * Lz));
  long rem = flat % ((long)Lz * Lz);
  int l = (int)(rem / Lz);
  int m = (int)(rem % Lz);
  const float* pf = pair_feat + flat * Pz;
  float acc[Hz];
#pragma unroll
  for (int h = 0; h < Hz; ++h) acc[h] = 0.f;
  for (int p = 0; p < Pz; ++p) {
    float v = pf[p];
    v = siluf(v);
#pragma unroll
    for (int h = 0; h < Hz; ++h) acc[h] += v * w[h * Pz + p];
  }
  float s = 0.f;
#pragma unroll
  for (int h = 0; h < Hz; ++h) {
    pfb[(((long)b * Hz + h) * Lz + l) * Lz + m] = acc[h];
    s += acc[h];
  }
  colscore[((long)b * Lz + m) * Lz + l] = s * (1.f / Hz);
}

// ---------------- top-3 rows per (b, column m); jax tie rule: lower index wins
struct Top3 { float v[3]; int i[3]; };
__device__ __forceinline__ void t3_insert(Top3& t, float v, int i) {
  bool b0 = (v > t.v[0]) || (v == t.v[0] && i < t.i[0]);
  bool b1 = (v > t.v[1]) || (v == t.v[1] && i < t.i[1]);
  bool b2 = (v > t.v[2]) || (v == t.v[2] && i < t.i[2]);
  if (b0) {
    t.v[2] = t.v[1]; t.i[2] = t.i[1];
    t.v[1] = t.v[0]; t.i[1] = t.i[0];
    t.v[0] = v; t.i[0] = i;
  } else if (b1) {
    t.v[2] = t.v[1]; t.i[2] = t.i[1];
    t.v[1] = v; t.i[1] = i;
  } else if (b2) {
    t.v[2] = v; t.i[2] = i;
  }
}

__global__ __launch_bounds__(128) void topk_kernel(const float* __restrict__ colscore,
                                                   int* __restrict__ idx3) {
  int bm = blockIdx.x;                       // b*L + m
  const float* col = colscore + (long)bm * Lz;
  int tid = threadIdx.x;
  Top3 t;
#pragma unroll
  for (int k = 0; k < 3; ++k) { t.v[k] = -3.4e38f; t.i[k] = 0x7fffffff; }
  for (int l = tid; l < Lz; l += 128) t3_insert(t, col[l], l);
  __shared__ float sv[128 * 3];
  __shared__ int   si[128 * 3];
#pragma unroll
  for (int k = 0; k < 3; ++k) { sv[tid * 3 + k] = t.v[k]; si[tid * 3 + k] = t.i[k]; }
  __syncthreads();
  if (tid == 0) {
    Top3 g;
#pragma unroll
    for (int k = 0; k < 3; ++k) { g.v[k] = -3.4e38f; g.i[k] = 0x7fffffff; }
    for (int j = 0; j < 128 * 3; ++j) t3_insert(g, sv[j], si[j]);
#pragma unroll
    for (int k = 0; k < 3; ++k) idx3[bm * 3 + k] = g.i[k];
  }
}

// ---------------- f2n[b,m,p] = keep[m] * mean_k pair_feat[b, idx_k, m, p]*keep[idx_k]
// written directly as f16 (only consumed by the pair2node GEMM)
__global__ __launch_bounds__(128) void f2n_kernel(const float* __restrict__ pair_feat,
                                                  const int* __restrict__ idx3,
                                                  const unsigned char* __restrict__ pad,
                                                  half_t* __restrict__ f2n) {
  int bm = blockIdx.x;
  int b = bm / Lz, m = bm % Lz;
  int p = threadIdx.x;
  float keep_m = pad[bm] ? 0.f : 1.f;
  float s = 0.f;
#pragma unroll
  for (int k = 0; k < 3; ++k) {
    int l = idx3[bm * 3 + k];
    float kl = pad[b * Lz + l] ? 0.f : 1.f;
    s += pair_feat[(((long)b * Lz + l) * Lz + m) * Pz + p] * kl;
  }
  f2n[(long)bm * Pz + p] = (half_t)(s * (1.f / 3.f) * keep_m);
}

// ---------------- cond = silu(x + f2n_mlp) (f16);  h0 = masked pos embedding (f32)
__global__ __launch_bounds__(256) void cond_pe_kernel(const float* __restrict__ x,
                                                      const float* __restrict__ f2nd,
                                                      const float* __restrict__ pos,
                                                      const float* __restrict__ pos_emb_w,
                                                      const unsigned char* __restrict__ pad,
                                                      half_t* __restrict__ cond,
                                                      float* __restrict__ hbuf) {
  long i = (long)blockIdx.x * 256 + threadIdx.x;   // b*L*D + l*D + d
  int d = (int)(i % Dz);
  long bl = i / Dz;
  int b = (int)(bl / Lz), l = (int)(bl % Lz);
  float xv = x[((long)l * Bz + b) * Dz + d] + f2nd[i];
  cond[i] = (half_t)siluf(xv);
  float keep = pad[bl] ? 0.f : 1.f;
  const float* pw = pos_emb_w + (long)d * 3;
  const float* pp = pos + bl * 3;
  hbuf[i] = keep * (pp[0] * pw[0] + pp[1] * pw[1] + pp[2] * pw[2]);
}

// ---------------- WMMA GEMM: out = act(A16[M,K] . W16[N,K]^T + bias)
// Wave computes a 16(M) x 64(N) strip; A fragment reused across 4 accumulators.
// Output to fp32 C or f16 Ch (exactly one non-null).
__global__ __launch_bounds__(256) void gemm16_kernel(const half_t* __restrict__ A,
                                                     const half_t* __restrict__ W,
                                                     const float* __restrict__ bias,
                                                     float* __restrict__ C,
                                                     half_t* __restrict__ Ch,
                                                     int M, int N, int K,
                                                     int lda, int ldw, int ldc,
                                                     int act) {
  int wave = threadIdx.x >> 5;
  int lane = threadIdx.x & 31;
  int half = lane >> 4;
  int r = lane & 15;
  int stripsN = N >> 6;                      // strips of 4 n-tiles
  int strip = blockIdx.x * 8 + wave;
  if (strip >= (M >> 4) * stripsN) return;
  int tm = strip / stripsN, ts = strip % stripsN;
  const half_t* arow = A + (long)(tm * 16 + r) * lda + half * 8;
  const half_t* wrow = W + (long)(ts * 64 + r) * ldw + half * 8;
  v8f acc0 = {}, acc1 = {}, acc2 = {}, acc3 = {};
  for (int k0 = 0; k0 < K; k0 += 32) {
    __builtin_prefetch(arow + k0 + 32, 0, 1);   // global_prefetch_b8: next A chunk
    v16h af = ld_frag(arow + k0);
    v16h b0 = ld_frag(wrow + k0);
    v16h b1 = ld_frag(wrow + k0 + 16 * ldw);
    v16h b2 = ld_frag(wrow + k0 + 32 * ldw);
    v16h b3 = ld_frag(wrow + k0 + 48 * ldw);
    acc0 = wmma_f16(af, b0, acc0);
    acc1 = wmma_f16(af, b1, acc1);
    acc2 = wmma_f16(af, b2, acc2);
    acc3 = wmma_f16(af, b3, acc3);
  }
  v8f accs[4] = {acc0, acc1, acc2, acc3};
#pragma unroll
  for (int j = 0; j < 4; ++j) {
    int n = ts * 64 + j * 16 + r;
    float bv = bias ? bias[n] : 0.f;
#pragma unroll
    for (int v = 0; v < 8; ++v) {
      int m = tm * 16 + v + 8 * half;
      float val = accs[j][v] + bv;
      if (act == 1) val = geluf(val);
      else if (act == 2) val = siluf(val);
      if (C) C[(long)m * ldc + n] = val;
      else   Ch[(long)m * ldc + n] = (half_t)val;
    }
  }
}

// ---------------- LN + adaLN modulate (optionally fuse residual gate first)
// hin_out (f32) = base (+te) (+gate*addsrc); a_out (f16) = LN(.)*(1+sc)+sh
__global__ __launch_bounds__(256) void ln_mod_kernel(const float* __restrict__ base,
                                                     const float* __restrict__ te,
                                                     const float* __restrict__ addsrc,
                                                     const float* __restrict__ mod,
                                                     int gate_off, int sh_off, int sc_off,
                                                     float* __restrict__ hin_out,
                                                     half_t* __restrict__ a_out) {
  int row = blockIdx.x;            // b*L + l
  int b = row / Lz;
  int tid = threadIdx.x;
  const float* mrow = mod + (long)row * 6 * Dz;
  __shared__ float red[256];
  float t[2];
  float s = 0.f, s2 = 0.f;
#pragma unroll
  for (int j = 0; j < 2; ++j) {
    int d = tid + j * 256;
    float v = base[(long)row * Dz + d];
    if (te) v += te[b * Dz + d];
    if (addsrc) v += mrow[gate_off + d] * addsrc[(long)row * Dz + d];
    t[j] = v; s += v; s2 += v * v;
  }
  red[tid] = s; __syncthreads();
  for (int o = 128; o > 0; o >>= 1) { if (tid < o) red[tid] += red[tid + o]; __syncthreads(); }
  s = red[0]; __syncthreads();
  red[tid] = s2; __syncthreads();
  for (int o = 128; o > 0; o >>= 1) { if (tid < o) red[tid] += red[tid + o]; __syncthreads(); }
  s2 = red[0];
  float mu = s * (1.f / Dz);
  float var = s2 * (1.f / Dz) - mu * mu;
  float rs = rsqrtf(var + EPSz);
#pragma unroll
  for (int j = 0; j < 2; ++j) {
    int d = tid + j * 256;
    hin_out[(long)row * Dz + d] = t[j];
    a_out[(long)row * Dz + d] =
        (half_t)((t[j] - mu) * rs * (1.f + mrow[sc_off + d]) + mrow[sh_off + d]);
  }
}

// ---------------- attention scores: s = q.k^T/sqrt(DH) + pfb, pad cols -> NEG (f32)
// Wave strip: 16 rows x 64 cols; single K-chunk (DH=32), q fragment reused 4x.
__global__ __launch_bounds__(256) void attn_score_kernel(const half_t* __restrict__ qkv,
                                                         const float* __restrict__ pfb,
                                                         const unsigned char* __restrict__ pad,
                                                         float* __restrict__ scores) {
  int z = blockIdx.y;              // b*H + h
  int b = z / Hz, h = z % Hz;
  int wave = threadIdx.x >> 5;
  int lane = threadIdx.x & 31;
  int half = lane >> 4, r = lane & 15;
  int strip = blockIdx.x * 8 + wave;     // 32 (M tiles) x 8 (N strips) = 256
  int tm = strip >> 3, ts = strip & 7;
  const half_t* qb = qkv + (long)b * Lz * 3 * Dz + h * DHz + half * 8;
  const half_t* kb = qkv + (long)b * Lz * 3 * Dz + Dz + h * DHz + half * 8;
  v16h af = ld_frag(qb + (long)(tm * 16 + r) * (3 * Dz));
  const half_t* krow = kb + (long)(ts * 64 + r) * (3 * Dz);
  v16h b0 = ld_frag(krow);
  v16h b1 = ld_frag(krow + 16 * (3 * Dz));
  v16h b2 = ld_frag(krow + 32 * (3 * Dz));
  v16h b3 = ld_frag(krow + 48 * (3 * Dz));
  v8f acc0 = {}, acc1 = {}, acc2 = {}, acc3 = {};
  acc0 = wmma_f16(af, b0, acc0);
  acc1 = wmma_f16(af, b1, acc1);
  acc2 = wmma_f16(af, b2, acc2);
  acc3 = wmma_f16(af, b3, acc3);
  v8f accs[4] = {acc0, acc1, acc2, acc3};
  const float scale = 0.17677669529663687f; // 1/sqrt(32)
  long base = (long)z * Lz * Lz;
#pragma unroll
  for (int j = 0; j < 4; ++j) {
    int n = ts * 64 + j * 16 + r;
    int padv = pad[b * Lz + n];
#pragma unroll
    for (int v = 0; v < 8; ++v) {
      int m = tm * 16 + v + 8 * half;
      float val = accs[j][v] * scale + pfb[base + (long)m * Lz + n];
      if (padv) val = NEGz;
      scores[base + (long)m * Lz + n] = val;
    }
  }
}

// ---------------- row softmax over 512 cols; reads f32 scores, emits f16 probs
__global__ __launch_bounds__(256) void softmax_kernel(const float* __restrict__ s,
                                                      half_t* __restrict__ p16) {
  long row = blockIdx.x;
  const float* p = s + row * Lz;
  half_t* q = p16 + row * Lz;
  int tid = threadIdx.x;
  __shared__ float red[256];
  float v0 = p[tid], v1 = p[tid + 256];
  red[tid] = fmaxf(v0, v1); __syncthreads();
  for (int o = 128; o > 0; o >>= 1) { if (tid < o) red[tid] = fmaxf(red[tid], red[tid + o]); __syncthreads(); }
  float mx = red[0]; __syncthreads();
  float e0 = __expf(v0 - mx), e1 = __expf(v1 - mx);
  red[tid] = e0 + e1; __syncthreads();
  for (int o = 128; o > 0; o >>= 1) { if (tid < o) red[tid] += red[tid + o]; __syncthreads(); }
  float inv = 1.f / red[0];
  q[tid] = (half_t)(e0 * inv);
  q[tid + 256] = (half_t)(e1 * inv);
}

// ---------------- transpose V out of qkv16: vT[(b*H+h)*DH + dh][m] = v[b,m,h,dh]
__global__ __launch_bounds__(256) void vtrans_kernel(const half_t* __restrict__ qkv,
                                                     half_t* __restrict__ vT) {
  long i = (long)blockIdx.x * 256 + threadIdx.x;   // over B*L*D
  long bm = i / Dz;
  int c = (int)(i % Dz);                            // h*DH + dh
  int b = (int)(bm / Lz), m = (int)(bm % Lz);
  int h = c / DHz, dh = c % DHz;
  vT[(((long)(b * Hz + h) * DHz + dh) * Lz) + m] = qkv[bm * 3 * Dz + 2 * Dz + c];
}

// ---------------- o16 = probs16 . V  (V pre-transposed to [z, DH, L] f16)
// Wave strip: 16 rows x 32 (= full DH) cols, 2 accumulators.
__global__ __launch_bounds__(256) void attn_pv_kernel(const half_t* __restrict__ probs,
                                                      const half_t* __restrict__ vT,
                                                      half_t* __restrict__ o) {
  int z = blockIdx.y;
  int b = z / Hz, h = z % Hz;
  int wave = threadIdx.x >> 5;
  int lane = threadIdx.x & 31;
  int half = lane >> 4, r = lane & 15;
  int tm = blockIdx.x * 8 + wave;       // 32 M-tiles
  const half_t* arow = probs + (long)z * Lz * Lz + (long)(tm * 16 + r) * Lz + half * 8;
  const half_t* w0 = vT + ((long)z * DHz + r) * Lz + half * 8;        // n = 0..15
  const half_t* w1 = vT + ((long)z * DHz + 16 + r) * Lz + half * 8;   // n = 16..31
  v8f acc0 = {}, acc1 = {};
  for (int k0 = 0; k0 < Lz; k0 += 32) {
    v16h af = ld_frag(arow + k0);
    v16h b0 = ld_frag(w0 + k0);
    v16h b1 = ld_frag(w1 + k0);
    acc0 = wmma_f16(af, b0, acc0);
    acc1 = wmma_f16(af, b1, acc1);
  }
  v8f accs[2] = {acc0, acc1};
#pragma unroll
  for (int j = 0; j < 2; ++j) {
    int n = j * 16 + r;
#pragma unroll
    for (int v = 0; v < 8; ++v) {
      int m = tm * 16 + v + 8 * half;
      o[(long)b * Lz * Dz + (long)m * Dz + h * DHz + n] = (half_t)accs[j][v];
    }
  }
}

// ---------------- out = hin + mod[gate_off] * src (f32)
__global__ __launch_bounds__(256) void gate_add_kernel(const float* __restrict__ hin,
                                                       const float* __restrict__ mod,
                                                       int gate_off,
                                                       const float* __restrict__ src,
                                                       float* __restrict__ out) {
  long i = (long)blockIdx.x * 256 + threadIdx.x;
  long row = i / Dz;
  int d = (int)(i % Dz);
  out[i] = hin[i] + mod[row * 6 * Dz + gate_off + d] * src[i];
}

// ---------------- host side ----------------
static void launch_gemm(const half_t* A, const half_t* W, const float* bias,
                        float* C, half_t* Ch,
                        int M, int N, int K, int lda, int ldw, int ldc,
                        int act, hipStream_t s) {
  int strips = (M / 16) * (N / 64);
  dim3 grid((strips + 7) / 8);
  gemm16_kernel<<<grid, 256, 0, s>>>(A, W, bias, C, Ch, M, N, K, lda, ldw, ldc, act);
}
static void launch_cvt(const float* src, half_t* dst, long n, hipStream_t s) {
  cvt16_kernel<<<dim3((unsigned)((n + 255) / 256)), 256, 0, s>>>(src, dst, n);
}

extern "C" void kernel_launch(void* const* d_in, const int* in_sizes, int n_in,
                              void* d_out, int out_size, void* d_ws, size_t ws_size,
                              hipStream_t stream) {
  (void)in_sizes; (void)n_in; (void)out_size; (void)ws_size;
  const float* x         = (const float*)d_in[0];   // [L,B,D]
  const float* pos       = (const float*)d_in[1];   // [B,L,3]
  const float* time_emb  = (const float*)d_in[2];   // [B,1,D]
  const float* pair_feat = (const float*)d_in[3];   // [B,L,L,P]
  const float* pos_emb_w = (const float*)d_in[4];   // [D,3]
  const float* pfb_w     = (const float*)d_in[5];   // [H,P]
  const float* p2n_w1    = (const float*)d_in[6];   // [P,P]
  const float* p2n_w2    = (const float*)d_in[7];   // [D,P]
  const float* adaln_w   = (const float*)d_in[8];   // [NL,6D,D]
  const float* adaln_b   = (const float*)d_in[9];   // [NL,6D]
  const float* qkv_w     = (const float*)d_in[10];  // [NL,3D,D]
  const float* qkv_b     = (const float*)d_in[11];  // [NL,3D]
  const float* proj_w    = (const float*)d_in[12];  // [NL,D,D]
  const float* proj_b    = (const float*)d_in[13];  // [NL,D]
  const float* mlp_w1    = (const float*)d_in[14];  // [NL,FF,D]
  const float* mlp_b1    = (const float*)d_in[15];  // [NL,FF]
  const float* mlp_w2    = (const float*)d_in[16];  // [NL,D,FF]
  const float* mlp_b2    = (const float*)d_in[17];  // [NL,D]
  const unsigned char* pad = (const unsigned char*)d_in[18]; // [B,L] bool bytes

  // workspace bump allocator
  char* wptr = (char*)d_ws;
  auto alloc = [&](size_t bytes) -> void* {
    void* p = (void*)wptr;
    wptr += (bytes + 255) & ~(size_t)255;
    return p;
  };
  const long BL  = (long)Bz * Lz;
  const long BLL = BL * Lz;
  // fp32 buffers
  float* pfb      = (float*)alloc(sizeof(float) * (size_t)Bz * Hz * Lz * Lz);
  float* scores   = (float*)alloc(sizeof(float) * (size_t)Bz * Hz * Lz * Lz);
  float* colscore = (float*)alloc(sizeof(float) * (size_t)BLL);
  int*   idx3     = (int*)  alloc(sizeof(int)   * (size_t)BL * 3);
  float* f2nd     = (float*)alloc(sizeof(float) * (size_t)BL * Dz);
  float* hbuf     = (float*)alloc(sizeof(float) * (size_t)BL * Dz);
  float* hin      = (float*)alloc(sizeof(float) * (size_t)BL * Dz);
  float* mod      = (float*)alloc(sizeof(float) * (size_t)BL * 6 * Dz);
  float* projo    = (float*)alloc(sizeof(float) * (size_t)BL * Dz);
  float* mlpo     = (float*)alloc(sizeof(float) * (size_t)BL * Dz);
  // f16 buffers
  half_t* probs16 = (half_t*)alloc(sizeof(half_t) * (size_t)Bz * Hz * Lz * Lz);
  half_t* f2n16   = (half_t*)alloc(sizeof(half_t) * (size_t)BL * Pz);
  half_t* t116    = (half_t*)alloc(sizeof(half_t) * (size_t)BL * Pz);
  half_t* cond16  = (half_t*)alloc(sizeof(half_t) * (size_t)BL * Dz);
  half_t* abuf16  = (half_t*)alloc(sizeof(half_t) * (size_t)BL * Dz);
  half_t* m_in16  = (half_t*)alloc(sizeof(half_t) * (size_t)BL * Dz);
  half_t* qkv16   = (half_t*)alloc(sizeof(half_t) * (size_t)BL * 3 * Dz);
  half_t* vT16    = (half_t*)alloc(sizeof(half_t) * (size_t)BL * Dz);
  half_t* obuf16  = (half_t*)alloc(sizeof(half_t) * (size_t)BL * Dz);
  half_t* mlph16  = (half_t*)alloc(sizeof(half_t) * (size_t)BL * FFz);
  // f16 weight arenas (converted once per call)
  half_t* adaln_w16 = (half_t*)alloc(sizeof(half_t) * (size_t)NLz * 6 * Dz * Dz);
  half_t* qkv_w16   = (half_t*)alloc(sizeof(half_t) * (size_t)NLz * 3 * Dz * Dz);
  half_t* proj_w16  = (half_t*)alloc(sizeof(half_t) * (size_t)NLz * Dz * Dz);
  half_t* mlp_w116  = (half_t*)alloc(sizeof(half_t) * (size_t)NLz * FFz * Dz);
  half_t* mlp_w216  = (half_t*)alloc(sizeof(half_t) * (size_t)NLz * Dz * FFz);
  half_t* p2n_w116  = (half_t*)alloc(sizeof(half_t) * (size_t)Pz * Pz);
  half_t* p2n_w216  = (half_t*)alloc(sizeof(half_t) * (size_t)Dz * Pz);

  // 0) one-shot fp32 -> f16 weight conversion
  launch_cvt(adaln_w, adaln_w16, (long)NLz * 6 * Dz * Dz, stream);
  launch_cvt(qkv_w,   qkv_w16,   (long)NLz * 3 * Dz * Dz, stream);
  launch_cvt(proj_w,  proj_w16,  (long)NLz * Dz * Dz,     stream);
  launch_cvt(mlp_w1,  mlp_w116,  (long)NLz * FFz * Dz,    stream);
  launch_cvt(mlp_w2,  mlp_w216,  (long)NLz * Dz * FFz,    stream);
  launch_cvt(p2n_w1,  p2n_w116,  (long)Pz * Pz,           stream);
  launch_cvt(p2n_w2,  p2n_w216,  (long)Dz * Pz,           stream);

  // 1) pair bias + column scores (streams pair_feat once)
  pfb_kernel<<<dim3((unsigned)(BLL / 256)), 256, 0, stream>>>(pair_feat, pfb_w, pfb, colscore);
  // 2) top-3 rows per column
  topk_kernel<<<dim3((unsigned)BL), 128, 0, stream>>>(colscore, idx3);
  // 3) masked gather + mean (f16 out)
  f2n_kernel<<<dim3((unsigned)BL), 128, 0, stream>>>(pair_feat, idx3, pad, f2n16);
  // 4) pair2node MLP: silu(f2n @ w1^T) @ w2^T
  launch_gemm(f2n16, p2n_w116, nullptr, nullptr, t116, (int)BL, Pz, Pz, Pz, Pz, Pz, 2, stream);
  launch_gemm(t116, p2n_w216, nullptr, f2nd, nullptr, (int)BL, Dz, Pz, Pz, Pz, Dz, 0, stream);
  // 5) cond = silu(x + f2nd) (f16); h0 = masked pos embedding (f32)
  cond_pe_kernel<<<dim3((unsigned)(BL * Dz / 256)), 256, 0, stream>>>(
      x, f2nd, pos, pos_emb_w, pad, cond16, hbuf);

  for (int l = 0; l < NLz; ++l) {
    // mod = cond @ adaln_w^T + adaln_b (f32)
    launch_gemm(cond16, adaln_w16 + (long)l * 6 * Dz * Dz, adaln_b + (long)l * 6 * Dz,
                mod, nullptr, (int)BL, 6 * Dz, Dz, Dz, Dz, 6 * Dz, 0, stream);
    // hin = h + time_emb (f32); a = LN(hin)*(1+sc_a)+sh_a (f16)
    ln_mod_kernel<<<dim3((unsigned)BL), 256, 0, stream>>>(
        hbuf, time_emb, nullptr, mod, 0, 0, Dz, hin, abuf16);
    // qkv = a @ qkv_w^T + qkv_b (f16)
    launch_gemm(abuf16, qkv_w16 + (long)l * 3 * Dz * Dz, qkv_b + (long)l * 3 * Dz,
                nullptr, qkv16, (int)BL, 3 * Dz, Dz, Dz, Dz, 3 * Dz, 0, stream);
    // attention
    attn_score_kernel<<<dim3(32, Bz * Hz), 256, 0, stream>>>(qkv16, pfb, pad, scores);
    softmax_kernel<<<dim3((unsigned)(Bz * Hz * Lz)), 256, 0, stream>>>(scores, probs16);
    vtrans_kernel<<<dim3((unsigned)(BL * Dz / 256)), 256, 0, stream>>>(qkv16, vT16);
    attn_pv_kernel<<<dim3(4, Bz * Hz), 256, 0, stream>>>(probs16, vT16, obuf16);
    // proj (f32 out, consumed by residual)
    launch_gemm(obuf16, proj_w16 + (long)l * Dz * Dz, proj_b + (long)l * Dz,
                projo, nullptr, (int)BL, Dz, Dz, Dz, Dz, Dz, 0, stream);
    // hin += g_a * projo (f32); m_in = LN(hin)*(1+sc_m)+sh_m (f16)
    ln_mod_kernel<<<dim3((unsigned)BL), 256, 0, stream>>>(
        hin, nullptr, projo, mod, 2 * Dz, 3 * Dz, 4 * Dz, hin, m_in16);
    // MLP
    launch_gemm(m_in16, mlp_w116 + (long)l * FFz * Dz, mlp_b1 + (long)l * FFz,
                nullptr, mlph16, (int)BL, FFz, Dz, Dz, Dz, FFz, 1, stream);
    launch_gemm(mlph16, mlp_w216 + (long)l * Dz * FFz, mlp_b2 + (long)l * Dz,
                mlpo, nullptr, (int)BL, Dz, FFz, FFz, FFz, Dz, 0, stream);
    // h = hin + g_m * mlpo (last layer writes the output)
    float* dest = (l == NLz - 1) ? (float*)d_out : hbuf;
    gate_add_kernel<<<dim3((unsigned)(BL * Dz / 256)), 256, 0, stream>>>(
        hin, mod, 5 * Dz, mlpo, dest);
  }
}
